// EnergyACM_49340584296529
// MI455X (gfx1250) — matrix operand order repaired and checked by reading, not compiled, not run on hardware
//
#include <hip/hip_runtime.h>
#include <stddef.h>

#define N_NODES 100000
#define IN_CH   256
#define HID     128
#define OUT_CH  64
#define N_LAYERS 4
#define MWAVE   5        // M-tiles per wave (6250 tiles = 1250 * 5)

typedef __attribute__((ext_vector_type(16))) __bf16          v16bf;
typedef __attribute__((ext_vector_type(8)))  float           v8f;
typedef __attribute__((ext_vector_type(8)))  unsigned short  u16x8;   // 16 bytes

// ---------- helpers ----------
__device__ __forceinline__ unsigned short f2bf_bits(float f) {
    unsigned u = __builtin_bit_cast(unsigned, f);
    u += 0x7FFFu + ((u >> 16) & 1u);          // round-to-nearest-even
    return (unsigned short)(u >> 16);
}
__device__ __forceinline__ __bf16 bits2bf(unsigned short s) {
    return __builtin_bit_cast(__bf16, s);
}
__device__ __forceinline__ v16bf frag_from_2x128(const unsigned short* p16) {
    // two contiguous 16-byte loads -> 16-element bf16 fragment
    u16x8 lo = *(const u16x8*)(p16);
    u16x8 hi = *(const u16x8*)(p16 + 8);
    v16bf r;
#pragma unroll
    for (int t = 0; t < 8; ++t) { r[t] = bits2bf(lo[t]); r[8 + t] = bits2bf(hi[t]); }
    return r;
}

// ---------- f32 -> bf16(bits), optional scale ----------
__global__ void cvt_f32_bf16_kernel(const float* __restrict__ src,
                                    unsigned short* __restrict__ dst,
                                    int n, float scale) {
    int i = blockIdx.x * blockDim.x + threadIdx.x;
    if (i >= n) return;
    dst[i] = f2bf_bits(src[i] * scale);
}

// ---------- pack weight (K x N row-major f32) into WMMA B-fragment order ----------
// out[((nt*KC + kc)*32 + lane)*16 + t] = bf16( W[(kc*32 + (lane>>4)*16 + t) * N + nt*16 + (lane&15)] )
__global__ void pack_b_kernel(const float* __restrict__ W,
                              unsigned short* __restrict__ out, int K, int N) {
    int i = blockIdx.x * blockDim.x + threadIdx.x;
    if (i >= K * N) return;
    int KC = K >> 5;
    int t    = i & 15;
    int lane = (i >> 4) & 31;
    int kc   = (i >> 9) % KC;
    int nt   = (i >> 9) / KC;
    int k = kc * 32 + (lane >> 4) * 16 + t;
    int n = nt * 16 + (lane & 15);
    out[i] = f2bf_bits(W[(size_t)k * N + n]);
}

// ---------- row pointers via binary search (edges sorted by row) ----------
__global__ void rowptr_kernel(const int* __restrict__ rows, int E,
                              int* __restrict__ row_ptr, int n) {
    int i = blockIdx.x * blockDim.x + threadIdx.x;
    if (i > n) return;
    int lo = 0, hi = E;
    while (lo < hi) {
        int mid = (lo + hi) >> 1;
        if (rows[mid] < i) lo = mid + 1; else hi = mid;
    }
    row_ptr[i] = lo;
}

__global__ void invdeg_kernel(const int* __restrict__ row_ptr,
                              float* __restrict__ invdeg, int n) {
    int i = blockIdx.x * blockDim.x + threadIdx.x;
    if (i >= n) return;
    int deg = row_ptr[i + 1] - row_ptr[i] + 1;   // +1 self loop
    invdeg[i] = 1.0f / (float)deg;
}

// ---------- WMMA bf16 GEMM: C = A(bf16,[M,K]) @ Bpacked(bf16) (+bias)(relu) ----------
// One wave computes MWAVE vertically-stacked 16x16 tiles of one N-tile column.
// M % (16*MWAVE) == 0, K % 32 == 0, N % 16 == 0. Whole-wave early exit keeps EXEC all-1.
__global__ void gemm_bf16_wmma_kernel(const unsigned short* __restrict__ A, int lda,
                                      const unsigned short* __restrict__ Bp,
                                      int K, int N,
                                      const float* __restrict__ bias,
                                      void* __restrict__ Cout, int c_bf16, int relu,
                                      int nwaves, int ntiles_n) {
    int gw = (int)((blockIdx.x * blockDim.x + threadIdx.x) >> 5);
    if (gw >= nwaves) return;
    int lane = threadIdx.x & 31;
    int grp = gw / ntiles_n;
    int nt  = gw - grp * ntiles_n;
    int mt0 = grp * MWAVE;
    int r15 = lane & 15;
    int hif = lane >> 4;
    int KC  = K >> 5;

    v8f c[MWAVE];
#pragma unroll
    for (int m = 0; m < MWAVE; ++m) c[m] = (v8f){};

    // A fragment base: lane row = mt0*16 + r15; chunk kc adds kc*32; 16 contiguous k at +hif*16
    const unsigned short* abase = A + (size_t)(mt0 * 16 + r15) * lda + hif * 16;
    // B packed: 32 bytes per lane per (nt, kc)
    const unsigned short* bbase = Bp + ((size_t)nt * KC * 32 + lane) * 16;

    for (int kc = 0; kc < KC; ++kc) {
        // ---- B fragment: two b128 loads (packed layout) ----
        const unsigned short* bp = bbase + (size_t)kc * 32 * 16;
        u16x8 blo = *(const u16x8*)(bp);
        u16x8 bhi = *(const u16x8*)(bp + 8);
        v16bf b;
#pragma unroll
        for (int t = 0; t < 8; ++t) { b[t] = bits2bf(blo[t]); b[8 + t] = bits2bf(bhi[t]); }

        const unsigned short* ap0 = abase + kc * 32;
#pragma unroll
        for (int m = 0; m < MWAVE; ++m) {
            const unsigned short* ap = ap0 + (size_t)m * 16 * lda;
            v16bf a = frag_from_2x128(ap);      // elems 0..15 at +0..+15 (k = 16*hif + s)
            c[m] = __builtin_amdgcn_wmma_f32_16x16x32_bf16(
                       false, a, false, b, (short)0, c[m], false, false);
        }
    }

    // C/D layout: VGPR r -> M = tile*16 + hif*8 + r, N = nt*16 + r15 (ldc == N)
    int col = nt * 16 + r15;
    float bv = bias ? bias[col] : 0.0f;
#pragma unroll
    for (int m = 0; m < MWAVE; ++m) {
        size_t cbase = (size_t)((mt0 + m) * 16 + hif * 8) * N + col;
#pragma unroll
        for (int r = 0; r < 8; ++r) {
            float v = c[m][r] + bv;
            if (relu) v = fmaxf(v, 0.0f);
            if (c_bf16) ((unsigned short*)Cout)[cbase + (size_t)r * N] = f2bf_bits(v);
            else        ((float*)Cout)[cbase + (size_t)r * N] = v;
        }
    }
}

// ---------- SPMM: y[i] = w_off(i) * sum_{j in adj(i)} x[j] + w_diag(i) * x[i] ----------
// mode 0 (A_rw^p): w_off = w_diag = invdeg^p
// mode 1 (L_rw^p): w_off = (-invdeg)^p, w_diag = (1-invdeg)^p
__global__ void spmm_kernel(const float* __restrict__ X, float* __restrict__ Y,
                            const int* __restrict__ row_ptr,
                            const int* __restrict__ col,
                            const float* __restrict__ invdeg,
                            int n, int mode, int p) {
    int w = (int)((blockIdx.x * blockDim.x + threadIdx.x) >> 5);
    int lane = threadIdx.x & 31;
    if (w >= n) return;
    int s = row_ptr[w], e = row_ptr[w + 1];
    const float4* Xv = (const float4*)X;
    float4 acc; acc.x = 0.f; acc.y = 0.f; acc.z = 0.f; acc.w = 0.f;
    for (int t = s; t < e; ++t) {
        int j = col[t];
        if (t + 4 < e) {                                   // gfx1250 global_prefetch path
            __builtin_prefetch(&Xv[(size_t)col[t + 4] * 32 + lane], 0, 1);
        }
        float4 v = Xv[(size_t)j * 32 + lane];
        acc.x += v.x; acc.y += v.y; acc.z += v.z; acc.w += v.w;
    }
    float id = invdeg[w];
    float boff  = (mode == 0) ? id : -id;
    float bdiag = (mode == 0) ? id : (1.0f - id);
    float woff = boff, wdiag = bdiag;
    for (int q = 1; q < p; ++q) { woff *= boff; wdiag *= bdiag; }
    float4 xd = Xv[(size_t)w * 32 + lane];
    float4 y;
    y.x = woff * acc.x + wdiag * xd.x;
    y.y = woff * acc.y + wdiag * xd.y;
    y.z = woff * acc.z + wdiag * xd.z;
    y.w = woff * acc.w + wdiag * xd.w;
    ((float4*)Y)[(size_t)w * 32 + lane] = y;
}

// ---------- acc (+)= relu(p) ----------
__global__ void acc_relu_kernel(float* __restrict__ acc, const float* __restrict__ p,
                                int n, int overwrite) {
    int i = blockIdx.x * blockDim.x + threadIdx.x;
    if (i >= n) return;
    float v = fmaxf(p[i], 0.0f);
    acc[i] = overwrite ? v : (acc[i] + v);
}

// ======================================================================
extern "C" void kernel_launch(void* const* d_in, const int* in_sizes, int n_in,
                              void* d_out, int out_size, void* d_ws, size_t ws_size,
                              hipStream_t stream) {
    const float* x      = (const float*)d_in[0];
    const float* W_lin1 = (const float*)d_in[1];
    const float* b_lin1 = (const float*)d_in[2];
    const float* W1     = (const float*)d_in[3];
    const float* W2     = (const float*)d_in[4];
    const float* W_lin2 = (const float*)d_in[5];
    const float* b_lin2 = (const float*)d_in[6];
    const int*   rows   = (const int*)d_in[7];
    const int E = in_sizes[7] / 2;
    const int* cols = rows + E;

    const int M = N_NODES;
    const size_t F32B = (size_t)M * HID * sizeof(float);          // 51.2 MB
    const size_t XBFB = (((size_t)M * IN_CH * 2) + 255) & ~(size_t)255;
    const size_t HBFB = (((size_t)M * HID * 2) + 255) & ~(size_t)255;

    char* ws = (char*)d_ws;
    size_t off = 0;
    float*          o     = (float*)(ws + off);          off += F32B;
    float*          tmp   = (float*)(ws + off);          off += F32B;
    float*          acc   = (float*)(ws + off);          off += F32B;
    unsigned short* xbf   = (unsigned short*)(ws + off); off += XBFB;
    unsigned short* hbf   = (unsigned short*)(ws + off); off += HBFB;
    unsigned short* accbf = (unsigned short*)(ws + off); off += HBFB;
    unsigned short* wpk   = (unsigned short*)(ws + off);
    const int WLIN1_OFF = 0;                  // 256*128
    const int W1_OFF    = 32768;              // 4*128*128
    const int W2_OFF    = 32768 + 4 * 16384;
    const int WLIN2_OFF = 32768 + 8 * 16384;  // 128*64
    const size_t WPK_BYTES = ((size_t)(WLIN2_OFF + HID * OUT_CH) * 2 + 255) & ~(size_t)255;
    off += WPK_BYTES;
    int*   row_ptr = (int*)(ws + off);        off += (((size_t)(M + 1) * 4) + 255) & ~(size_t)255;
    float* invdeg  = (float*)(ws + off);

    // --- convert x to bf16; pack weights into WMMA B-fragment order ---
    {
        int nx = M * IN_CH;
        cvt_f32_bf16_kernel<<<(nx + 255) / 256, 256, 0, stream>>>(x, xbf, nx, 1.0f);
        int n1 = IN_CH * HID;
        pack_b_kernel<<<(n1 + 255) / 256, 256, 0, stream>>>(W_lin1, wpk + WLIN1_OFF, IN_CH, HID);
        int nl = HID * HID;
        for (int l = 0; l < N_LAYERS; ++l) {
            pack_b_kernel<<<(nl + 255) / 256, 256, 0, stream>>>(
                W1 + (size_t)l * nl, wpk + W1_OFF + l * nl, HID, HID);
            pack_b_kernel<<<(nl + 255) / 256, 256, 0, stream>>>(
                W2 + (size_t)l * nl, wpk + W2_OFF + l * nl, HID, HID);
        }
        int n3 = HID * OUT_CH;
        pack_b_kernel<<<(n3 + 255) / 256, 256, 0, stream>>>(W_lin2, wpk + WLIN2_OFF, HID, OUT_CH);
    }

    // --- CSR row pointers + inverse degrees (edges sorted by row) ---
    rowptr_kernel<<<(M + 1 + 255) / 256, 256, 0, stream>>>(rows, E, row_ptr, M);
    invdeg_kernel<<<(M + 255) / 256, 256, 0, stream>>>(row_ptr, invdeg, M);

    auto gemm = [&](const unsigned short* A, int lda, const unsigned short* Bp,
                    int K, int N, const float* bias, void* C, int c_bf16, int relu) {
        int ntn = N / 16;
        int nwaves = (M / (16 * MWAVE)) * ntn;
        int blocks = (nwaves + 7) / 8;                    // 8 waves (256 threads) / block
        gemm_bf16_wmma_kernel<<<blocks, 256, 0, stream>>>(A, lda, Bp, K, N, bias, C,
                                                          c_bf16, relu, nwaves, ntn);
    };
    const int spmm_blocks = (M * 32 + 255) / 256;
    const int HN = M * HID;
    auto spmm = [&](const float* X, float* Y, int mode, int p) {
        spmm_kernel<<<spmm_blocks, 256, 0, stream>>>(X, Y, row_ptr, cols, invdeg, M, mode, p);
    };

    // --- h = relu(x @ W_lin1 + b_lin1), stored directly as bf16 ---
    gemm(xbf, IN_CH, wpk + WLIN1_OFF, IN_CH, HID, b_lin1, hbf, 1, 1);

    // --- layers ---
    for (int l = 0; l < N_LAYERS; ++l) {
        for (int path = 0; path < 2; ++path) {   // path0 -> W1/o1, path1 -> W2/o2
            const unsigned short* W = wpk + (path == 0 ? W1_OFF : W2_OFF) + l * HID * HID;
            gemm(hbf, HID, W, HID, HID, nullptr, o, 0, 0);

            int nops, modes[2], pows[2];
            if (l == 0)                 { nops = 1; modes[0] = path;     pows[0] = 1; }
            else if (l == N_LAYERS - 1) { nops = 1; modes[0] = 1 - path; pows[0] = l; }
            else { nops = 2; modes[0] = 1 - path; pows[0] = l; modes[1] = path; pows[1] = 1; }

            float* cur = o; float* nxt = tmp;
            for (int k2 = 0; k2 < nops; ++k2) {
                spmm(cur, nxt, modes[k2], pows[k2]);
                float* t = cur; cur = nxt; nxt = t;
            }
            acc_relu_kernel<<<(HN + 255) / 256, 256, 0, stream>>>(
                acc, cur, HN, (l == 0 && path == 0) ? 1 : 0);
        }
    }

    // --- out = (acc / 8) @ W_lin2 + b_lin2 ---
    cvt_f32_bf16_kernel<<<(HN + 255) / 256, 256, 0, stream>>>(
        acc, accbf, HN, 1.0f / (2 * N_LAYERS));
    gemm(accbf, HID, wpk + WLIN2_OFF, HID, OUT_CH, b_lin2, d_out, 0, 0);
}
